// ScaledDotProductAttention_19988777796223
// MI455X (gfx1250) — matrix operand order repaired and checked by reading, not compiled, not run on hardware
//
#include <hip/hip_runtime.h>

typedef __attribute__((ext_vector_type(16))) __bf16 v16bf;
typedef __attribute__((ext_vector_type(8)))  float  v8f;

#define B_    2
#define H_    16
#define S_    2048
#define D_    64
#define BR_   64            // query rows per workgroup (16 per wave, 4 waves)
#define BC_   64            // keys per tile
#define NT_   (S_ / BC_)    // 32 key tiles
#define NWAVE 4
#define LP_   72            // padded LDS pitch: 144B rows -> 16B-aligned + conflict-free
// 1/sqrt(64) * log2(e): softmax in exp2 domain (v_exp_f32 is base-2 native)
#define QSCALE_ (0.125f * 1.44269504088896f)

// ---------------- prep kernels (write bf16 K and bf16 V^T into d_ws) -------------

__global__ __launch_bounds__(256) void conv_k_bf16(const float* __restrict__ src,
                                                   __bf16* __restrict__ dst) {
    const size_t i = ((size_t)blockIdx.x * 256 + threadIdx.x) * 4;
    const float4 f = *(const float4*)(src + i);
    dst[i + 0] = (__bf16)f.x; dst[i + 1] = (__bf16)f.y;
    dst[i + 2] = (__bf16)f.z; dst[i + 3] = (__bf16)f.w;
}

// V[bh][s][d] (f32) -> V^T[bh][d][s] (bf16), 64x64 tiles through LDS
__global__ __launch_bounds__(256) void transp_v_bf16(const float* __restrict__ v,
                                                     __bf16* __restrict__ vt) {
    __shared__ float t[64 * 65];
    const int bh = blockIdx.x >> 5;
    const int st = blockIdx.x & 31;                 // 64-row s-tile
    const float* src = v + (size_t)bh * S_ * D_ + (size_t)st * 64 * D_;
    __bf16*      dst = vt + (size_t)bh * D_ * S_ + (size_t)st * 64;
#pragma unroll
    for (int j = 0; j < 16; ++j) {
        const int idx = j * 256 + threadIdx.x;      // s*64 + d, coalesced read
        t[(idx >> 6) * 65 + (idx & 63)] = src[idx];
    }
    __syncthreads();
#pragma unroll
    for (int j = 0; j < 16; ++j) {
        const int idx = j * 256 + threadIdx.x;      // d*64 + s, coalesced write
        const int d = idx >> 6, s = idx & 63;
        dst[(size_t)d * S_ + s] = (__bf16)t[s * 65 + d];
    }
}

// ---------------- flash-attention forward -----------------------------------------
// PRE=true : K/V^T already bf16 in global; async-DMA straight into padded LDS tiles.
// PRE=false: self-contained fallback (async f32 staging + in-kernel convert/transpose).

template <bool PRE>
__global__ __launch_bounds__(128) void fa_fwd_gfx1250(const float* __restrict__ q,
                                                      const float* __restrict__ kf32,
                                                      const float* __restrict__ vf32,
                                                      const __bf16* __restrict__ kbf,
                                                      const __bf16* __restrict__ vtbf,
                                                      float* __restrict__ out) {
    __shared__ float  Kf[PRE ? 1 : BC_ * D_];       // f32 staging (fallback only)
    __shared__ float  Vf[PRE ? 1 : BC_ * D_];
    __shared__ __bf16 Kt[BC_ * LP_];                // [key][d], pitch 72
    __shared__ __bf16 VtT[D_ * LP_];                // [d][key], pitch 72
    __shared__ __bf16 Pt[NWAVE][16 * LP_];          // per-wave P tile [row][key]

    const int tid  = threadIdx.x;
    const int wave = tid >> 5;
    const int lane = tid & 31;
    const int l16  = lane & 15;
    const int hi   = lane >> 4;

    const int nqb = S_ / BR_;
    const int bh  = blockIdx.x / nqb;
    const int qb  = blockIdx.x % nqb;

    const float* qp = q + (size_t)bh * S_ * D_;
    float*       op = out + (size_t)bh * S_ * D_;

    const unsigned ktBase = (unsigned)(size_t)&Kt[0];
    const unsigned vtBase = (unsigned)(size_t)&VtT[0];
    const unsigned kfBase = (unsigned)(size_t)&Kf[0];
    const unsigned vfBase = (unsigned)(size_t)&Vf[0];

    // async-DMA one tile (PRE: bf16 -> padded compute tiles; else f32 -> staging)
    auto issue_tile = [&](int kt) {
        if constexpr (PRE) {
            const __bf16* kb  = kbf  + (size_t)bh * S_ * D_ + (size_t)kt * BC_ * D_;
            const __bf16* vtb = vtbf + (size_t)bh * D_ * S_ + (size_t)kt * BC_;
#pragma unroll
            for (int g = 0; g < 4; ++g) {
                const int e = (tid * 4 + g) * 8;            // linear half index (8 halfs/granule)
                const int r0 = e >> 6, c0 = e & 63;         // K: (key,d)   V^T: (d,key)
                const unsigned lk = ktBase + (unsigned)(r0 * LP_ + c0) * 2;
                const unsigned gk = (unsigned)e * 2;        // K rows contiguous in global
                asm volatile("global_load_async_to_lds_b128 %0, %1, %2"
                             :: "v"(lk), "v"(gk), "s"(kb) : "memory");
                const unsigned lv = vtBase + (unsigned)(r0 * LP_ + c0) * 2;
                const unsigned gv = (unsigned)(r0 * S_ + c0) * 2;  // V^T row pitch = S_
                asm volatile("global_load_async_to_lds_b128 %0, %1, %2"
                             :: "v"(lv), "v"(gv), "s"(vtb) : "memory");
            }
        } else {
            const float* kn = kf32 + (size_t)bh * S_ * D_ + (size_t)kt * BC_ * D_;
            const float* vn = vf32 + (size_t)bh * S_ * D_ + (size_t)kt * BC_ * D_;
#pragma unroll
            for (int i = 0; i < 8; ++i) {
                const unsigned voff = (unsigned)((i * 512 + tid * 4) * 4);
                const unsigned lk = kfBase + voff;
                const unsigned lv = vfBase + voff;
                asm volatile("global_load_async_to_lds_b128 %0, %1, %2"
                             :: "v"(lk), "v"(voff), "s"(kn) : "memory");
                asm volatile("global_load_async_to_lds_b128 %0, %1, %2"
                             :: "v"(lv), "v"(voff), "s"(vn) : "memory");
            }
        }
    };

    // ---- Q rows -> WMMA A-layout (pre-scaled into exp2 domain) ----
    // A operand as packed here (16x32): M = lane%16 ; K = h + 16*hi (contiguous k, same order as B)
    const int qrow = qb * BR_ + wave * 16 + l16;
    v16bf aq0, aq1, vones;
#pragma unroll
    for (int h = 0; h < 16; ++h) {
        const int dk = h + 16 * hi;
        aq0[h] = (__bf16)(qp[(size_t)qrow * D_ + dk]      * QSCALE_);
        aq1[h] = (__bf16)(qp[(size_t)qrow * D_ + 32 + dk] * QSCALE_);
        vones[h] = (__bf16)1.0f;
    }

    v8f o[4] = {v8f{}, v8f{}, v8f{}, v8f{}};
    float m[8], l[8];
#pragma unroll
    for (int r = 0; r < 8; ++r) { m[r] = -1e30f; l[r] = 0.0f; }

    issue_tile(0);   // prologue

    for (int kt = 0; kt < NT_; ++kt) {
        // ---- wait for this tile's async DMAs, make visible WG-wide ----
        asm volatile("s_wait_asynccnt 0x0" ::: "memory");
        __syncthreads();

        if constexpr (!PRE) {
            // fallback: convert f32 staging -> bf16 tiles (K straight, V transposed)
#pragma unroll
            for (int i = 0; i < 8; ++i) {
                const int fi  = i * 512 + tid * 4;   // key*64 + d
                const int key = fi >> 6, d = fi & 63;
                const float4 kv = *(const float4*)&Kf[fi];
                __bf16* kd = &Kt[key * LP_ + d];
                kd[0] = (__bf16)kv.x; kd[1] = (__bf16)kv.y;
                kd[2] = (__bf16)kv.z; kd[3] = (__bf16)kv.w;
                const float4 vv = *(const float4*)&Vf[fi];
                VtT[(d + 0) * LP_ + key] = (__bf16)vv.x;
                VtT[(d + 1) * LP_ + key] = (__bf16)vv.y;
                VtT[(d + 2) * LP_ + key] = (__bf16)vv.z;
                VtT[(d + 3) * LP_ + key] = (__bf16)vv.w;
            }
            __syncthreads();
        }

        // ---- overlap: kick off next tile's DMA while we compute ----
        if (kt + 1 < NT_) issue_tile(kt + 1);

        // ---- S = Q K^T : four 16-key blocks x two 32-deep chunks ----
        // B operand as packed here (32x16): N = lane%16 (key), K = h + 16*hi (d in chunk, same order as A)
        v8f s[4];
#pragma unroll
        for (int blk = 0; blk < 4; ++blk) {
            const int key = blk * 16 + l16;
            v16bf b0, b1;
#pragma unroll
            for (int h = 0; h < 16; ++h) {
                const int d0 = h + 16 * hi;
                b0[h] = Kt[key * LP_ + d0];
                b1[h] = Kt[key * LP_ + 32 + d0];
            }
            v8f acc = {};
            acc = __builtin_amdgcn_wmma_f32_16x16x32_bf16(false, aq0, false, b0,
                                                          (short)0, acc, false, false);
            acc = __builtin_amdgcn_wmma_f32_16x16x32_bf16(false, aq1, false, b1,
                                                          (short)0, acc, false, false);
            s[blk] = acc;
        }

        // ---- online softmax over 64 keys (exp2 domain) ----
        // C layout: lane holds rows r+8*hi (r=0..7), col = lane%16;
        // row max = xor butterfly within each 16-lane half (wave32-safe)
        float p[4][8], alpha[8];
#pragma unroll
        for (int r = 0; r < 8; ++r) {
            float sm = fmaxf(fmaxf(s[0][r], s[1][r]), fmaxf(s[2][r], s[3][r]));
#pragma unroll
            for (int off = 8; off >= 1; off >>= 1)
                sm = fmaxf(sm, __shfl_xor(sm, off, 32));
            const float nm = fmaxf(m[r], sm);
#pragma unroll
            for (int blk = 0; blk < 4; ++blk)
                p[blk][r] = exp2f(s[blk][r] - nm);
            alpha[r] = exp2f(m[r] - nm);
            m[r] = nm;
        }
#pragma unroll
        for (int r = 0; r < 8; ++r) {
            o[0][r] *= alpha[r]; o[1][r] *= alpha[r];
            o[2][r] *= alpha[r]; o[3][r] *= alpha[r];
        }

        // ---- P (C-layout) -> per-wave LDS -> A-layout; wave-private, DS in-order ----
#pragma unroll
        for (int r = 0; r < 8; ++r) {
            const int row = r + 8 * hi;
#pragma unroll
            for (int blk = 0; blk < 4; ++blk)
                Pt[wave][row * LP_ + blk * 16 + l16] = (__bf16)p[blk][r];
        }
        asm volatile("" ::: "memory");   // keep compiler from hoisting the reads

        v16bf ap0, ap1;
#pragma unroll
        for (int h = 0; h < 16; ++h) {
            const int kk = h + 16 * hi;
            ap0[h] = Pt[wave][l16 * LP_ + kk];
            ap1[h] = Pt[wave][l16 * LP_ + 32 + kk];
        }

        // ---- row sums via WMMA (P @ ones): replaces the sum butterfly ----
        v8f rsv = {};
        rsv = __builtin_amdgcn_wmma_f32_16x16x32_bf16(false, ap0, false, vones,
                                                      (short)0, rsv, false, false);
        rsv = __builtin_amdgcn_wmma_f32_16x16x32_bf16(false, ap1, false, vones,
                                                      (short)0, rsv, false, false);
#pragma unroll
        for (int r = 0; r < 8; ++r)
            l[r] = l[r] * alpha[r] + rsv[r];

        // ---- O += P @ V : per 16-wide d block, two 32-deep chunks ----
#pragma unroll
        for (int n = 0; n < 4; ++n) {
            v16bf bv0, bv1;
            const int drow = (n * 16 + l16) * LP_;
#pragma unroll
            for (int h = 0; h < 16; ++h) {
                const int kx = h + 16 * hi;
                bv0[h] = VtT[drow + kx];
                bv1[h] = VtT[drow + 32 + kx];
            }
            o[n] = __builtin_amdgcn_wmma_f32_16x16x32_bf16(false, ap0, false, bv0,
                                                           (short)0, o[n], false, false);
            o[n] = __builtin_amdgcn_wmma_f32_16x16x32_bf16(false, ap1, false, bv1,
                                                           (short)0, o[n], false, false);
        }
        // loop-top wait+barrier separates these reads from next tile's LDS writes
    }

    // ---- epilogue: normalize rows, store fp32 (coalesced per half-wave) ----
#pragma unroll
    for (int r = 0; r < 8; ++r) {
        const float inv = 1.0f / l[r];
        const int row = qb * BR_ + wave * 16 + r + 8 * hi;
#pragma unroll
        for (int n = 0; n < 4; ++n)
            op[(size_t)row * D_ + n * 16 + l16] = o[n][r] * inv;
    }
}

extern "C" void kernel_launch(void* const* d_in, const int* in_sizes, int n_in,
                              void* d_out, int out_size, void* d_ws, size_t ws_size,
                              hipStream_t stream) {
    (void)in_sizes; (void)n_in; (void)out_size;
    const float* q = (const float*)d_in[0];
    const float* k = (const float*)d_in[1];
    const float* v = (const float*)d_in[2];
    float* out = (float*)d_out;

    const size_t nelem = (size_t)B_ * H_ * S_ * D_;        // 4,194,304
    const size_t need  = nelem * 2 * 2;                    // bf16 K + bf16 V^T = 16 MB

    dim3 fgrid(B_ * H_ * (S_ / BR_));                      // 1024 workgroups
    dim3 fblock(128);                                      // 4 wave32s

    if (ws_size >= need) {
        __bf16* kb  = (__bf16*)d_ws;
        __bf16* vtb = kb + nelem;
        conv_k_bf16<<<dim3((unsigned)(nelem / 1024)), 256, 0, stream>>>(k, kb);
        transp_v_bf16<<<dim3(B_ * H_ * (S_ / 64)), 256, 0, stream>>>(v, vtb);
        fa_fwd_gfx1250<true><<<fgrid, fblock, 0, stream>>>(q, k, v, kb, vtb, out);
    } else {
        fa_fwd_gfx1250<false><<<fgrid, fblock, 0, stream>>>(q, k, v, nullptr, nullptr, out);
    }
}
